// EncoderRNN_61005715472777
// MI455X (gfx1250) — matrix operand order repaired and checked by reading, not compiled, not run on hardware
//
#include <hip/hip_runtime.h>
#include <hip/hip_bf16.h>
#include <math.h>

// Bidirectional GRU, S=2048, B=32, E=H=512.
// Persistent recurrent kernel: 2 directions x 16 hidden-slices = 32 WGs.
// Whh gate fragments live in REGISTERS for all 2048 steps (384 VGPRs/lane),
// Wih slice lives in LDS (bf16), x_t/h_t staged by async global->LDS DMA,
// GEMMs via v_wmma_f32_16x16x32_bf16 with fp32 accumulation.

#define S_LEN 2048
#define BATCH 32
#define EMB   512
#define HID   512
#define NSLICE 16          // workgroups per direction
#define HS     32          // hidden units per WG
#define NROWS  96          // 3 gates * HS weight rows per WG
#define KDIM   512         // E == H == 512

typedef __attribute__((ext_vector_type(16))) __bf16 v16bf;
typedef __attribute__((ext_vector_type(8)))  float  v8f;
typedef __attribute__((ext_vector_type(4)))  __bf16 bf4;
typedef int v4i_vs __attribute__((vector_size(16)));   // matches async-LDS builtin param

#if __has_builtin(__builtin_amdgcn_global_load_async_to_lds_b128) && \
    __has_builtin(__builtin_amdgcn_s_wait_asynccnt)
#define USE_ASYNC_LDS 1
typedef __attribute__((address_space(1))) v4i_vs* gptr_b128;
typedef __attribute__((address_space(3))) v4i_vs* lptr_b128;
#else
#define USE_ASYNC_LDS 0
#endif

__device__ __forceinline__ float sigmoidf_(float x) {
  return 1.0f / (1.0f + __expf(-x));
}

__global__ void gru_init_kernel(float* hbufs, unsigned* bar) {
  int i = blockIdx.x * blockDim.x + threadIdx.x;
  if (i < 2 * 2 * BATCH * HID) hbufs[i] = 0.0f;
  if (i < 2) bar[i] = 0u;
}

__global__ __launch_bounds__(128, 1)
void gru_persistent_kernel(const float* __restrict__ x,     // [S,B,E]
                           const float* __restrict__ wih_f, const float* __restrict__ whh_f,
                           const float* __restrict__ bih_f, const float* __restrict__ bhh_f,
                           const float* __restrict__ wih_b, const float* __restrict__ whh_b,
                           const float* __restrict__ bih_b, const float* __restrict__ bhh_b,
                           float* __restrict__ out,          // [B, 2H]
                           float* __restrict__ hbufs,        // [2 dir][2 parity][B*H]
                           unsigned* __restrict__ bar)       // [2]
{
  // 224 KB static LDS (gfx1250: up to 320 KB per workgroup)
  __shared__ __bf16 s_wih[NROWS * KDIM];   // 96 KB: Wih slice (bf16)
  __shared__ float  s_x32[BATCH * KDIM];   // 64 KB: x_t (fp32, DMA target)
  __shared__ float  s_h32[BATCH * KDIM];   // 64 KB: h_t (fp32, DMA target)

  const int tid   = threadIdx.x;
  const int lane  = tid & 31;
  const int wave  = tid >> 5;            // 4 waves
  const int slice = blockIdx.x;          // 0..15
  const int dir   = blockIdx.y;          // 0 fwd, 1 bwd
  const int j0    = slice * HS;

  const float* Wih = dir ? wih_b : wih_f;
  const float* Whh = dir ? whh_b : whh_f;
  const float* bih = dir ? bih_b : bih_f;
  const float* bhh = dir ? bhh_b : bhh_f;

  // ---- stage Wih slice fp32 -> bf16 into LDS ----
  for (int i = tid; i < NROWS * (KDIM / 4); i += 128) {
    const int row = i >> 7;        // 128 float4 per row
    const int c4  = i & 127;
    const int g   = row >> 5;
    const int rr  = row & 31;
    const float4 a = ((const float4*)(Wih + (size_t)(g * HID + j0 + rr) * KDIM))[c4];
    bf4 pa = { (__bf16)a.x, (__bf16)a.y, (__bf16)a.z, (__bf16)a.w };
    *(bf4*)(s_wih + row * KDIM + c4 * 4) = pa;
  }

  // Wave -> tile mapping: mt = batch tile (0..1), ng = hidden half (0..1).
  const int mt    = wave >> 1;
  const int ng    = wave & 1;
  const int khalf = lane >> 4;                 // K-half select (this code's contiguous order: k = 16*khalf + s)
  const int jl    = ng * 16 + (lane & 15);     // hidden unit within slice (0..31)
  const int jg    = j0 + jl;                   // global hidden unit (0..511)
  const int arow  = mt * 16 + (lane & 15);     // A-matrix row (batch)

  // lane-resident biases (gate order r,z,n; bhh_n stays inside the r* term)
  const float br_x = bih[jg],           br_h = bhh[jg];
  const float bz_x = bih[HID + jg],     bz_h = bhh[HID + jg];
  const float bn_x = bih[2 * HID + jg], bn_h = bhh[2 * HID + jg];

  // ---- hoist Whh B-fragments into registers for all 2048 steps ----
  // B frag (32x16): lane n holds weight row (g*HID + jg), K[khalf*16 .. +15]
  v16bf whhf[48];                       // 3 gates x 16 k-steps = 384 VGPRs
#pragma unroll
  for (int g = 0; g < 3; ++g) {
    const float* rp = Whh + ((size_t)g * HID + jg) * KDIM + khalf * 16;
#pragma unroll
    for (int k = 0; k < 16; ++k) {
      const float* p = rp + k * 32;
      const float4 f0 = *(const float4*)(p);
      const float4 f1 = *(const float4*)(p + 4);
      const float4 f2 = *(const float4*)(p + 8);
      const float4 f3 = *(const float4*)(p + 12);
      v16bf v;
      v[0]  = (__bf16)f0.x; v[1]  = (__bf16)f0.y; v[2]  = (__bf16)f0.z; v[3]  = (__bf16)f0.w;
      v[4]  = (__bf16)f1.x; v[5]  = (__bf16)f1.y; v[6]  = (__bf16)f1.z; v[7]  = (__bf16)f1.w;
      v[8]  = (__bf16)f2.x; v[9]  = (__bf16)f2.y; v[10] = (__bf16)f2.z; v[11] = (__bf16)f2.w;
      v[12] = (__bf16)f3.x; v[13] = (__bf16)f3.y; v[14] = (__bf16)f3.z; v[15] = (__bf16)f3.w;
      whhf[g * 16 + k] = v;
    }
  }

  // A frag (16x32 bf16) built from fp32 LDS: row arow, K[khalf*16..+15]
  auto load_a32 = [&](const float* base, int k) -> v16bf {
    const float* p = base + arow * KDIM + k * 32 + khalf * 16;
    const float4 f0 = *(const float4*)(p);
    const float4 f1 = *(const float4*)(p + 4);
    const float4 f2 = *(const float4*)(p + 8);
    const float4 f3 = *(const float4*)(p + 12);
    v16bf v;
    v[0]  = (__bf16)f0.x; v[1]  = (__bf16)f0.y; v[2]  = (__bf16)f0.z; v[3]  = (__bf16)f0.w;
    v[4]  = (__bf16)f1.x; v[5]  = (__bf16)f1.y; v[6]  = (__bf16)f1.z; v[7]  = (__bf16)f1.w;
    v[8]  = (__bf16)f2.x; v[9]  = (__bf16)f2.y; v[10] = (__bf16)f2.z; v[11] = (__bf16)f2.w;
    v[12] = (__bf16)f3.x; v[13] = (__bf16)f3.y; v[14] = (__bf16)f3.z; v[15] = (__bf16)f3.w;
    return v;
  };
  // Wih B frag from bf16 LDS: lane n holds K[khalf*16 .. +15] contiguous
  auto load_b = [&](int g, int k) -> v16bf {
    union { v16bf v; uint4 u[2]; } f;
    const __bf16* p = s_wih + (g * HS + jl) * KDIM + k * 32 + khalf * 16;
    f.u[0] = *(const uint4*)p;
    f.u[1] = *(const uint4*)(p + 8);
    return f.v;
  };

  float* hb0 = hbufs + (size_t)dir * 2 * BATCH * HID;
  unsigned* cnt = bar + dir;

  __syncthreads();

  for (int t = 0; t < S_LEN; ++t) {
    const int par = t & 1;
    const float* hread  = hb0 + par * BATCH * HID;
    float*       hwrite = hb0 + (1 - par) * BATCH * HID;
    const int teff = dir ? (S_LEN - 1 - t) : t;
    const float* xt = x + (size_t)teff * BATCH * EMB;

    // ---- stage x_t and h_t (fp32) into LDS ----
#if USE_ASYNC_LDS
    for (int i = tid; i < BATCH * (KDIM / 4); i += 128) {
      __builtin_amdgcn_global_load_async_to_lds_b128(
          (gptr_b128)(xt + i * 4), (lptr_b128)(s_x32 + i * 4), 0, 0);
      __builtin_amdgcn_global_load_async_to_lds_b128(
          (gptr_b128)(hread + i * 4), (lptr_b128)(s_h32 + i * 4), 0, 0);
    }
    __builtin_amdgcn_s_wait_asynccnt(0);
#else
    for (int i = tid; i < BATCH * (KDIM / 4); i += 128) {
      ((float4*)s_x32)[i] = ((const float4*)xt)[i];
      ((float4*)s_h32)[i] = ((const float4*)hread)[i];
    }
#endif
    __syncthreads();

    // two GEMMs: gx = x_t @ WihSlice^T (LDS B), gh = h_t @ WhhSlice^T (reg B)
    v8f ax_r = {}, ax_z = {}, ax_n = {};
    v8f ah_r = {}, ah_z = {}, ah_n = {};
#pragma unroll
    for (int k = 0; k < KDIM / 32; ++k) {
      const v16bf fah = load_a32(s_h32, k);
      ah_r = __builtin_amdgcn_wmma_f32_16x16x32_bf16(false, fah, false, whhf[k],      (short)0, ah_r, false, false);
      ah_z = __builtin_amdgcn_wmma_f32_16x16x32_bf16(false, fah, false, whhf[16 + k], (short)0, ah_z, false, false);
      ah_n = __builtin_amdgcn_wmma_f32_16x16x32_bf16(false, fah, false, whhf[32 + k], (short)0, ah_n, false, false);
      const v16bf fax = load_a32(s_x32, k);
      const v16bf wr  = load_b(0, k);
      const v16bf wz  = load_b(1, k);
      const v16bf wn  = load_b(2, k);
      ax_r = __builtin_amdgcn_wmma_f32_16x16x32_bf16(false, fax, false, wr, (short)0, ax_r, false, false);
      ax_z = __builtin_amdgcn_wmma_f32_16x16x32_bf16(false, fax, false, wz, (short)0, ax_z, false, false);
      ax_n = __builtin_amdgcn_wmma_f32_16x16x32_bf16(false, fax, false, wn, (short)0, ax_n, false, false);
    }

    // GRU gate math (PyTorch order r,z,n) + h update
#pragma unroll
    for (int i = 0; i < 8; ++i) {
      const int b = mt * 16 + i + khalf * 8;        // C/D layout: lanes 16-31 -> M+8
      const float r  = sigmoidf_(ax_r[i] + br_x + ah_r[i] + br_h);
      const float z  = sigmoidf_(ax_z[i] + bz_x + ah_z[i] + bz_h);
      const float nn = tanhf(ax_n[i] + bn_x + r * (ah_n[i] + bn_h));
      const float hold = s_h32[b * HID + jg];
      const float hnew = (1.0f - z) * nn + z * hold;
      hwrite[b * HID + jg] = hnew;
      if (t == S_LEN - 1) out[b * (2 * HID) + dir * HID + jg] = hnew;
    }

    // prefetch next timestep's x while we wait at the barrier
    if (t + 1 < S_LEN) {
      const int tn = dir ? (S_LEN - 2 - t) : (t + 1);
      const char* xn = (const char*)(x + (size_t)tn * BATCH * EMB);
      __builtin_prefetch(xn + tid * 512, 0, 1);     // 128 thr * 512B = 64 KB
    }

    // cross-workgroup recurrence barrier (per direction, monotonic counter)
    __threadfence();
    __syncthreads();
    if (tid == 0) {
      __hip_atomic_fetch_add(cnt, 1u, __ATOMIC_RELEASE, __HIP_MEMORY_SCOPE_AGENT);
      const unsigned target = (unsigned)(NSLICE * (t + 1));
      while (__hip_atomic_load(cnt, __ATOMIC_ACQUIRE, __HIP_MEMORY_SCOPE_AGENT) < target) {
        __builtin_amdgcn_s_sleep(2);
      }
    }
    __syncthreads();
  }
}

extern "C" void kernel_launch(void* const* d_in, const int* in_sizes, int n_in,
                              void* d_out, int out_size, void* d_ws, size_t ws_size,
                              hipStream_t stream) {
  (void)in_sizes; (void)n_in; (void)out_size; (void)ws_size;
  const float* x     = (const float*)d_in[1];   // d_in[0] = seq_length scalar (==2048)
  const float* wih_f = (const float*)d_in[2];
  const float* whh_f = (const float*)d_in[3];
  const float* bih_f = (const float*)d_in[4];
  const float* bhh_f = (const float*)d_in[5];
  const float* wih_b = (const float*)d_in[6];
  const float* whh_b = (const float*)d_in[7];
  const float* bih_b = (const float*)d_in[8];
  const float* bhh_b = (const float*)d_in[9];
  float* out = (float*)d_out;

  float* hbufs  = (float*)d_ws;                                  // 2*2*32*512 floats
  unsigned* bar = (unsigned*)((char*)d_ws + (size_t)2 * 2 * BATCH * HID * sizeof(float));

  const int ninit = 2 * 2 * BATCH * HID;
  gru_init_kernel<<<(ninit + 255) / 256, 256, 0, stream>>>(hbufs, bar);

  dim3 grid(NSLICE, 2);
  gru_persistent_kernel<<<grid, 128, 0, stream>>>(x, wih_f, whh_f, bih_f, bhh_f,
                                                  wih_b, whh_b, bih_b, bhh_b,
                                                  out, hbufs, bar);
}
